// JPT1_5823975653889
// MI455X (gfx1250) — matrix operand order repaired and checked by reading, not compiled, not run on hardware
//
#include <hip/hip_runtime.h>
#include <cstdint>
#include <cstddef>

// Problem constants (match reference)
#define SEQ   2048
#define EMB   512
#define NH    8
#define NL    4
#define FFD   2048
#define VOC   32000
#define BATCH 2
#define DH    64
#define MTOT  (BATCH*SEQ)   // 4096

typedef __attribute__((ext_vector_type(16))) __bf16 v16bf;
typedef __attribute__((ext_vector_type(8)))  float  v8f;

struct U4x2 { uint4 a, b; };

__device__ __forceinline__ unsigned short f2b(float f) {
  unsigned int u = __float_as_uint(f);
  unsigned int r = u + 0x7FFFu + ((u >> 16) & 1u);   // round-to-nearest-even
  return (unsigned short)(r >> 16);
}
__device__ __forceinline__ float b2f(unsigned short h) {
  return __uint_as_float(((unsigned int)h) << 16);
}

// Build a 16xbf16 fragment from two 16B chunks (no unions -> SROA-friendly).
__device__ __forceinline__ v16bf ld_frag(const unsigned short* p0,
                                         const unsigned short* p1) {
  U4x2 t;
  t.a = *(const uint4*)p0;
  t.b = *(const uint4*)p1;
  return __builtin_bit_cast(v16bf, t);
}

__device__ __forceinline__ v8f wmma_bf16(v16bf a, v16bf b, v8f c) {
  return __builtin_amdgcn_wmma_f32_16x16x32_bf16(false, a, false, b, (short)0, c,
                                                 false, false);
}

__device__ __forceinline__ v8f vzero8() {
  v8f z = {0.f, 0.f, 0.f, 0.f, 0.f, 0.f, 0.f, 0.f};
  return z;
}

// CDNA5 async global->LDS copy (ASYNCcnt path). The LDS destination VGPR holds
// the AS(3) byte offset == low 32 bits of the generic pointer to __shared__.
__device__ __forceinline__ void async_ld_lds_b128(void* ldsPtr, const void* g) {
  unsigned off = (unsigned)(uintptr_t)ldsPtr;
  asm volatile("global_load_async_to_lds_b128 %0, %1, off"
               :: "v"(off), "v"(g)
               : "memory");
}
__device__ __forceinline__ void wait_async0() {
  asm volatile("s_wait_asynccnt 0x0" ::: "memory");
}

// ---------------------------------------------------------------------------
// Weight convert + transpose: w[K][N] f32  ->  wT[N][K] bf16
// ---------------------------------------------------------------------------
__global__ __launch_bounds__(256) void convT_kernel(const float* __restrict__ w,
                                                    unsigned short* __restrict__ wT,
                                                    int K, int N) {
  size_t idx = (size_t)blockIdx.x * 256 + threadIdx.x;
  int n = (int)(idx % N);
  int k = (int)(idx / N);
  wT[(size_t)n * K + k] = f2b(w[idx]);
}

// ---------------------------------------------------------------------------
// Embedding: h[b,s,:] = tok_emb[x[b,s],:] + pos_emb[s,:]
// ---------------------------------------------------------------------------
__global__ __launch_bounds__(256) void embed_kernel(const int* __restrict__ x,
                                                    const float* __restrict__ tok,
                                                    const float* __restrict__ pos,
                                                    float* __restrict__ h) {
  size_t idx = (size_t)blockIdx.x * 256 + threadIdx.x;  // over MTOT*EMB
  int e = (int)(idx % EMB);
  int m = (int)(idx / EMB);
  int s = m % SEQ;
  h[idx] = tok[(size_t)x[m] * EMB + e] + pos[(size_t)s * EMB + e];
}

// ---------------------------------------------------------------------------
// LayerNorm over E=512, output bf16 (GEMM A operand). One block per row.
// ---------------------------------------------------------------------------
__global__ __launch_bounds__(256) void ln_kernel(const float* __restrict__ h,
                                                 const float* __restrict__ w,
                                                 const float* __restrict__ b,
                                                 unsigned short* __restrict__ y) {
  __shared__ float s1[256], s2[256];
  int row = blockIdx.x, t = threadIdx.x;
  const float* hr = h + (size_t)row * EMB;
  float a0 = hr[t], a1 = hr[t + 256];
  s1[t] = a0 + a1;
  s2[t] = a0 * a0 + a1 * a1;
  __syncthreads();
  for (int o = 128; o > 0; o >>= 1) {
    if (t < o) { s1[t] += s1[t + o]; s2[t] += s2[t + o]; }
    __syncthreads();
  }
  float mean = s1[0] * (1.0f / EMB);
  float var  = s2[0] * (1.0f / EMB) - mean * mean;
  float inv  = rsqrtf(var + 1e-5f);
  unsigned short* yr = y + (size_t)row * EMB;
  yr[t]       = f2b((a0 - mean) * inv * w[t]       + b[t]);
  yr[t + 256] = f2b((a1 - mean) * inv * w[t + 256] + b[t + 256]);
}

// ---------------------------------------------------------------------------
// WMMA GEMM: C[M,N] = A[M,K](bf16) @ BT[N,K](bf16)^T + bias, with epilogues.
//   mode 0: QKV scatter -> q [BH,S,D]*scale, k [BH,S,D], v transposed [BH,D,S]
//   mode 1: outF = resF + val              (residual add, fp32)
//   mode 2: outB = bf16(gelu_exact(val))   (FF activation)
//   mode 3: outF = val                     (logits)
// Workgroup tile 128x128, 8 waves (4 in M x 2 in N), wave tile 32x64, K-step 32.
// Double-buffered LDS staging via CDNA5 async global->LDS copies.
// ---------------------------------------------------------------------------
#define LDA 40   // LDS stride (bf16 elems): 80B rows -> conflict-free frag reads

__device__ __forceinline__ void epilogue_elem(
    float val, int rowg, int colg, int N, int mode, float qscale,
    float* __restrict__ outF, const float* __restrict__ resF,
    unsigned short* __restrict__ outB,
    unsigned short* __restrict__ qOut, unsigned short* __restrict__ kOut,
    unsigned short* __restrict__ vtOut) {
  if (mode == 0) {
    int sec = colg >> 9;             // 0=q 1=k 2=v (E=512)
    int rem = colg & 511;
    int hh = rem >> 6, d = rem & 63;
    int bb = rowg >> 11;             // SEQ=2048
    int s  = rowg & 2047;
    size_t bh = (size_t)bb * NH + hh;
    if (sec == 0)      qOut[(bh * SEQ + s) * DH + d] = f2b(val * qscale);
    else if (sec == 1) kOut[(bh * SEQ + s) * DH + d] = f2b(val);
    else               vtOut[(bh * DH + d) * SEQ + s] = f2b(val);
  } else if (mode == 1) {
    size_t idx = (size_t)rowg * N + colg;
    outF[idx] = resF[idx] + val;
  } else if (mode == 2) {
    float g = 0.5f * val * (1.0f + erff(val * 0.70710678118f));
    outB[(size_t)rowg * N + colg] = f2b(g);
  } else {
    outF[(size_t)rowg * N + colg] = val;
  }
}

__global__ __launch_bounds__(256) void gemm_kernel(
    const unsigned short* __restrict__ A, const unsigned short* __restrict__ BT,
    const float* __restrict__ bias, int M, int N, int K, int mode, float qscale,
    float* __restrict__ outF, const float* __restrict__ resF,
    unsigned short* __restrict__ outB,
    unsigned short* __restrict__ qOut, unsigned short* __restrict__ kOut,
    unsigned short* __restrict__ vtOut) {
  __shared__ unsigned short As[2][128 * LDA];
  __shared__ unsigned short Bs[2][128 * LDA];

  const int t = threadIdx.x;
  const int lane = t & 31, wave = t >> 5;
  const int lh = lane >> 4, lc = lane & 15;
  const int wm = wave & 3, wn = wave >> 2;
  const int m0 = blockIdx.y * 128, n0 = blockIdx.x * 128;

  // Individually named accumulators (keeps everything in VGPRs).
  v8f acc00 = vzero8(), acc01 = vzero8(), acc02 = vzero8(), acc03 = vzero8();
  v8f acc10 = vzero8(), acc11 = vzero8(), acc12 = vzero8(), acc13 = vzero8();

  const int lr  = t >> 1;              // 0..127: tile row
  const int lc2 = (t & 1) << 4;        // 0 or 16: half of 32-wide K slab
  const unsigned short* gA = A  + (size_t)(m0 + lr) * K + lc2;
  const unsigned short* gB = BT + (size_t)(n0 + lr) * K + lc2;
  const int soff = lr * LDA + lc2;     // staging offset within a buffer

  // Fragment base offsets within LDS tiles
  const int arow0 = (wm * 32 + lc) * LDA;           // A rows (mi=0); +16*LDA for mi=1
  const int bcol0 = (wn * 64 + lc) * LDA;           // B cols (ni=0); +16*LDA per ni

#define ISSUE(BUF, KOFF)                                                     \
  {                                                                          \
    async_ld_lds_b128(&As[BUF][soff],     gA + (KOFF));                      \
    async_ld_lds_b128(&As[BUF][soff + 8], gA + (KOFF) + 8);                  \
    async_ld_lds_b128(&Bs[BUF][soff],     gB + (KOFF));                      \
    async_ld_lds_b128(&Bs[BUF][soff + 8], gB + (KOFF) + 8);                  \
  }

  ISSUE(0, 0)
  int cur = 0;
  for (int k0 = 0; k0 < K; k0 += 32) {
    wait_async0();        // current tile landed in LDS (this wave's copies)
    __syncthreads();      // ...and everyone else's; prior reads of other buf done
    if (k0 + 32 < K) ISSUE(cur ^ 1, k0 + 32)   // prefetch next tile, other buffer

    const unsigned short* Ac = &As[cur][0];
    const unsigned short* Bc = &Bs[cur][0];
    // A fragments: row = lane%16; K = lh*16 + e (contiguous)
    v16bf fa0 = ld_frag(&Ac[arow0 + lh * 16],            &Ac[arow0 + lh * 16 + 8]);
    v16bf fa1 = ld_frag(&Ac[arow0 + 16 * LDA + lh * 16], &Ac[arow0 + 16 * LDA + lh * 16 + 8]);
    // B fragments: col = lane%16; K = lh*16 + e (contiguous in BT layout)
    v16bf fb0 = ld_frag(&Bc[bcol0 + lh * 16],            &Bc[bcol0 + lh * 16 + 8]);
    v16bf fb1 = ld_frag(&Bc[bcol0 + 16 * LDA + lh * 16], &Bc[bcol0 + 16 * LDA + lh * 16 + 8]);
    v16bf fb2 = ld_frag(&Bc[bcol0 + 32 * LDA + lh * 16], &Bc[bcol0 + 32 * LDA + lh * 16 + 8]);
    v16bf fb3 = ld_frag(&Bc[bcol0 + 48 * LDA + lh * 16], &Bc[bcol0 + 48 * LDA + lh * 16 + 8]);

    acc00 = wmma_bf16(fa0, fb0, acc00);
    acc01 = wmma_bf16(fa0, fb1, acc01);
    acc02 = wmma_bf16(fa0, fb2, acc02);
    acc03 = wmma_bf16(fa0, fb3, acc03);
    acc10 = wmma_bf16(fa1, fb0, acc10);
    acc11 = wmma_bf16(fa1, fb1, acc11);
    acc12 = wmma_bf16(fa1, fb2, acc12);
    acc13 = wmma_bf16(fa1, fb3, acc13);
    cur ^= 1;
  }
#undef ISSUE

  // Epilogue. C layout: row = base + r + 8*lh, col = base + lane%16.
  const int rb = m0 + wm * 32 + 8 * lh;
  const int cb = n0 + wn * 64 + lc;
#define EPI(ACC, MI, NI)                                                        \
  {                                                                             \
    int colg = cb + (NI) * 16;                                                  \
    float bv = bias[colg];                                                      \
    _Pragma("unroll") for (int r = 0; r < 8; ++r) {                             \
      epilogue_elem(ACC[r] + bv, rb + (MI) * 16 + r, colg, N, mode, qscale,     \
                    outF, resF, outB, qOut, kOut, vtOut);                       \
    }                                                                           \
  }
  EPI(acc00, 0, 0) EPI(acc01, 0, 1) EPI(acc02, 0, 2) EPI(acc03, 0, 3)
  EPI(acc10, 1, 0) EPI(acc11, 1, 1) EPI(acc12, 1, 2) EPI(acc13, 1, 3)
#undef EPI
}

// ---------------------------------------------------------------------------
// Causal attention, one workgroup per (16-query tile, head, batch).
// q,k: [B*H, S, D] bf16 (q pre-scaled by 1/sqrt(D)); vt: [B*H, D, S] bf16.
// Scores kept as bf16 in LDS (16 x 2056), softmax in LDS, P@V via WMMA with
// deterministic per-wave partial reduction. o: [B,S,E] bf16.
// ---------------------------------------------------------------------------
#define SPLD 2056   // score row stride (bf16): conflict-free A-frag reads

__global__ __launch_bounds__(256) void attn_kernel(
    const unsigned short* __restrict__ q, const unsigned short* __restrict__ k,
    const unsigned short* __restrict__ vt, unsigned short* __restrict__ o) {
  extern __shared__ unsigned char smem_raw[];
  unsigned short* sp  = (unsigned short*)smem_raw;          // [16][SPLD]
  float* partials = (float*)(smem_raw + 16 * SPLD * 2);     // [8][16][64]
  float* red      = partials + 8 * 16 * 64;                 // [16][16]
  float* rowsum   = red + 256;                              // [16]

  const int qt = blockIdx.x;           // query tile (16 rows)
  const int hh = blockIdx.y;
  const int bz = blockIdx.z;
  const int bh = bz * NH + hh;
  const int q0 = qt * 16;
  const int t = threadIdx.x;
  const int wave = t >> 5, lane = t & 31;
  const int lh = lane >> 4, lc = lane & 15;

  const unsigned short* qbase = q  + (size_t)bh * SEQ * DH;
  const unsigned short* kbase = k  + (size_t)bh * SEQ * DH;
  const unsigned short* vbase = vt + (size_t)bh * DH * SEQ;

  // ---- Phase 1: S = Q K^T (scale folded into q), causal mask, -> LDS bf16
  const unsigned short* qr = qbase + (size_t)(q0 + lc) * DH;
  v16bf qf0 = ld_frag(qr + lh * 16,      qr + lh * 16 + 8);
  v16bf qf1 = ld_frag(qr + 32 + lh * 16, qr + 40 + lh * 16);

  for (int j = wave; j <= qt; j += 8) {          // key tiles (causal: j<=qt)
    const unsigned short* kr0 = kbase + (size_t)(j * 16 + lc) * DH + lh * 16;
    v16bf kf0 = ld_frag(kr0,      kr0 + 8);
    v16bf kf1 = ld_frag(kr0 + 32, kr0 + 40);
    v8f c = vzero8();
    c = wmma_bf16(qf0, kf0, c);
    c = wmma_bf16(qf1, kf1, c);
    int col = j * 16 + lc;
#pragma unroll
    for (int r = 0; r < 8; ++r) {
      int lrow = r + 8 * lh;
      float val = c[r];
      if (col > q0 + lrow) val = -1e30f;
      sp[lrow * SPLD + col] = f2b(val);
    }
  }
  __syncthreads();

  // ---- Phase 2: softmax (unnormalized exp kept; rowsum divided at the end)
  const int nj = qt + 1;
  const int ncols = nj * 16;
  {
    int rowi = t >> 4, tr = t & 15;
    unsigned short* rowp = sp + rowi * SPLD;
    int c0 = tr * nj, c1 = c0 + nj;
    float lmax = -1e30f;
    for (int cI = c0; cI < c1; ++cI) lmax = fmaxf(lmax, b2f(rowp[cI]));
    red[rowi * 16 + tr] = lmax;
    __syncthreads();
    float rmax = -1e30f;
    for (int i = 0; i < 16; ++i) rmax = fmaxf(rmax, red[rowi * 16 + i]);
    __syncthreads();
    float lsum = 0.f;
    for (int cI = c0; cI < c1; ++cI) {
      float pe = __expf(b2f(rowp[cI]) - rmax);
      rowp[cI] = f2b(pe);
      lsum += pe;
    }
    red[rowi * 16 + tr] = lsum;
    __syncthreads();
    float rs = 0.f;
    for (int i = 0; i < 16; ++i) rs += red[rowi * 16 + i];
    if (tr == 0) rowsum[rowi] = rs;
    if (nj & 1) sp[rowi * SPLD + ncols + tr] = 0;   // zero-pad to K%32==0
  }
  __syncthreads();

  // ---- Phase 3: O_partial = P @ V  (per-wave K-chunks, WMMA)
  const int kchunks = (nj + 1) >> 1;               // 32-key chunks
  {
    v8f a0 = vzero8(), a1 = vzero8(), a2 = vzero8(), a3 = vzero8();
    for (int c = wave; c < kchunks; c += 8) {
      const unsigned short* pr = sp + lc * SPLD + c * 32;
      v16bf pf = ld_frag(pr + lh * 16, pr + lh * 16 + 8);
      const unsigned short* vr0 = vbase + (size_t)lc * SEQ + c * 32 + lh * 16;
      v16bf vf0 = ld_frag(vr0,                vr0 + 8);
      v16bf vf1 = ld_frag(vr0 + 16 * SEQ,     vr0 + 16 * SEQ + 8);
      v16bf vf2 = ld_frag(vr0 + 32 * SEQ,     vr0 + 32 * SEQ + 8);
      v16bf vf3 = ld_frag(vr0 + 48 * SEQ,     vr0 + 48 * SEQ + 8);
      a0 = wmma_bf16(pf, vf0, a0);
      a1 = wmma_bf16(pf, vf1, a1);
      a2 = wmma_bf16(pf, vf2, a2);
      a3 = wmma_bf16(pf, vf3, a3);
    }
#define STP(ACC, NI)                                                           \
    _Pragma("unroll") for (int r = 0; r < 8; ++r)                              \
      partials[(wave * 16 + (r + 8 * lh)) * 64 + (NI) * 16 + lc] = ACC[r];
    STP(a0, 0) STP(a1, 1) STP(a2, 2) STP(a3, 3)
#undef STP
  }
  __syncthreads();

  // ---- Phase 4: deterministic reduce over 8 waves, divide by rowsum, store
#pragma unroll
  for (int i = 0; i < 4; ++i) {
    int idx = t * 4 + i;                 // 1024 = 16 rows x 64 cols
    int row = idx >> 6, d = idx & 63;
    float s = 0.f;
#pragma unroll
    for (int w2 = 0; w2 < 8; ++w2) s += partials[(w2 * 16 + row) * 64 + d];
    s /= rowsum[row];
    size_t m = (size_t)bz * SEQ + q0 + row;
    o[m * EMB + hh * DH + d] = f2b(s);
  }
}

// ---------------------------------------------------------------------------
// Host orchestration
// ---------------------------------------------------------------------------
extern "C" void kernel_launch(void* const* d_in, const int* in_sizes, int n_in,
                              void* d_out, int out_size, void* d_ws, size_t ws_size,
                              hipStream_t stream) {
  (void)in_sizes; (void)n_in; (void)out_size; (void)ws_size;
  const int*   x       = (const int*)  d_in[0];
  const float* tok_emb = (const float*)d_in[1];
  const float* pos_emb = (const float*)d_in[2];
  const float* ln1_w   = (const float*)d_in[3];
  const float* ln1_b   = (const float*)d_in[4];
  const float* qkv_w   = (const float*)d_in[5];
  const float* qkv_b   = (const float*)d_in[6];
  const float* proj_w  = (const float*)d_in[7];
  const float* proj_b  = (const float*)d_in[8];
  const float* ln2_w   = (const float*)d_in[9];
  const float* ln2_b   = (const float*)d_in[10];
  const float* ff1_w   = (const float*)d_in[11];
  const float* ff1_b   = (const float*)d_in[12];
  const float* ff2_w   = (const float*)d_in[13];
  const float* ff2_b   = (const float*)d_in[14];
  const float* lnf_w   = (const float*)d_in[15];
  const float* lnf_b   = (const float*)d_in[16];
  const float* out_w   = (const float*)d_in[17];
  const float* out_b   = (const float*)d_in[18];
  float* out = (float*)d_out;

  // Workspace carve
  char* p = (char*)d_ws;
  auto alloc = [&](size_t bytes) -> char* {
    char* r = p;
    p += (bytes + 255) & ~(size_t)255;
    return r;
  };
  float*          h     = (float*)         alloc((size_t)MTOT * EMB * 4);
  unsigned short* abuf  = (unsigned short*)alloc((size_t)MTOT * EMB * 2);   // LN out / attn out
  unsigned short* act   = (unsigned short*)alloc((size_t)MTOT * FFD * 2);   // GELU activations
  unsigned short* qb    = (unsigned short*)alloc((size_t)BATCH * NH * SEQ * DH * 2);
  unsigned short* kb    = (unsigned short*)alloc((size_t)BATCH * NH * SEQ * DH * 2);
  unsigned short* vtb   = (unsigned short*)alloc((size_t)BATCH * NH * DH * SEQ * 2);
  unsigned short* qkvT  = (unsigned short*)alloc((size_t)NL * 3 * EMB * EMB * 2);
  unsigned short* projT = (unsigned short*)alloc((size_t)NL * EMB * EMB * 2);
  unsigned short* ff1T  = (unsigned short*)alloc((size_t)NL * FFD * EMB * 2);
  unsigned short* ff2T  = (unsigned short*)alloc((size_t)NL * EMB * FFD * 2);
  unsigned short* outT  = (unsigned short*)alloc((size_t)VOC * EMB * 2);

  const float qscale = 0.125f;   // 1/sqrt(DH)
  const size_t attn_smem = (size_t)16 * SPLD * 2 + 8 * 16 * 64 * 4 + 256 * 4 + 16 * 4;

  // 1) Convert + transpose all weights to bf16 [N,K]
  convT_kernel<<<(EMB * (size_t)VOC) / 256, 256, 0, stream>>>(out_w, outT, EMB, VOC);
  for (int l = 0; l < NL; ++l) {
    convT_kernel<<<((size_t)EMB * 3 * EMB) / 256, 256, 0, stream>>>(
        qkv_w + (size_t)l * EMB * 3 * EMB, qkvT + (size_t)l * 3 * EMB * EMB, EMB, 3 * EMB);
    convT_kernel<<<((size_t)EMB * EMB) / 256, 256, 0, stream>>>(
        proj_w + (size_t)l * EMB * EMB, projT + (size_t)l * EMB * EMB, EMB, EMB);
    convT_kernel<<<((size_t)EMB * FFD) / 256, 256, 0, stream>>>(
        ff1_w + (size_t)l * EMB * FFD, ff1T + (size_t)l * FFD * EMB, EMB, FFD);
    convT_kernel<<<((size_t)FFD * EMB) / 256, 256, 0, stream>>>(
        ff2_w + (size_t)l * FFD * EMB, ff2T + (size_t)l * EMB * FFD, FFD, EMB);
  }

  // 2) Embedding
  embed_kernel<<<((size_t)MTOT * EMB) / 256, 256, 0, stream>>>(x, tok_emb, pos_emb, h);

  // 3) Transformer layers
  for (int l = 0; l < NL; ++l) {
    ln_kernel<<<MTOT, 256, 0, stream>>>(h, ln1_w + (size_t)l * EMB, ln1_b + (size_t)l * EMB, abuf);

    gemm_kernel<<<dim3((3 * EMB) / 128, MTOT / 128), 256, 0, stream>>>(
        abuf, qkvT + (size_t)l * 3 * EMB * EMB, qkv_b + (size_t)l * 3 * EMB,
        MTOT, 3 * EMB, EMB, /*mode=*/0, qscale,
        nullptr, nullptr, nullptr, qb, kb, vtb);

    attn_kernel<<<dim3(SEQ / 16, NH, BATCH), 256, attn_smem, stream>>>(qb, kb, vtb, abuf);

    gemm_kernel<<<dim3(EMB / 128, MTOT / 128), 256, 0, stream>>>(
        abuf, projT + (size_t)l * EMB * EMB, proj_b + (size_t)l * EMB,
        MTOT, EMB, EMB, /*mode=*/1, 0.f,
        h, h, nullptr, nullptr, nullptr, nullptr);

    ln_kernel<<<MTOT, 256, 0, stream>>>(h, ln2_w + (size_t)l * EMB, ln2_b + (size_t)l * EMB, abuf);

    gemm_kernel<<<dim3(FFD / 128, MTOT / 128), 256, 0, stream>>>(
        abuf, ff1T + (size_t)l * FFD * EMB, ff1_b + (size_t)l * FFD,
        MTOT, FFD, EMB, /*mode=*/2, 0.f,
        nullptr, nullptr, act, nullptr, nullptr, nullptr);

    gemm_kernel<<<dim3(EMB / 128, MTOT / 128), 256, 0, stream>>>(
        act, ff2T + (size_t)l * EMB * FFD, ff2_b + (size_t)l * EMB,
        MTOT, EMB, FFD, /*mode=*/1, 0.f,
        h, h, nullptr, nullptr, nullptr, nullptr);
  }

  // 4) Final LN + LM head
  ln_kernel<<<MTOT, 256, 0, stream>>>(h, lnf_w, lnf_b, abuf);
  gemm_kernel<<<dim3(VOC / 128, MTOT / 128), 256, 0, stream>>>(
      abuf, outT, out_b, MTOT, VOC, EMB, /*mode=*/3, 0.f,
      out, nullptr, nullptr, nullptr, nullptr, nullptr);
}
